// EdgeEmbeddingModel_26113401160366
// MI455X (gfx1250) — matrix operand order repaired and checked by reading, not compiled, not run on hardware
//
#include <hip/hip_runtime.h>
#include <hip/hip_bf16.h>
#include <math.h>

// ---------------------------------------------------------------------------
// Problem constants (match reference)
// ---------------------------------------------------------------------------
#define NN      65536      // nodes per graph
#define F_IN    32
#define DD      64         // edge embedding dim
#define HH_     2          // heads
#define OUT1    128        // gat1 out feats per head
#define HID     128
#define BB      128        // graphs
#define NEDGES  512
#define SLOPE   0.2f
#define IN_SIZE 65536      // NEDGES * DD * HH_

// ---------------------------------------------------------------------------
// WMMA types
// ---------------------------------------------------------------------------
typedef __attribute__((ext_vector_type(16))) __bf16 v16bf;
typedef __attribute__((ext_vector_type(8)))  __bf16 v8bf;
typedef __attribute__((ext_vector_type(4)))  __bf16 v4bf;
typedef __attribute__((ext_vector_type(8)))  float  v8f;

enum { GF_BIAS = 1, GF_RELU = 2, GF_ACC = 4 };

// ---------------------------------------------------------------------------
// Generic GEMM: C(M,N) = [ACC? C +] A(M,K) @ B(K,N) [+ bias] [relu]
// fp32 operands converted to bf16 in LDS; fp32 accumulate via
// v_wmma_f32_16x16x32_bf16.  Block = 256 threads = 8 waves.
// Tile: BM=64, BN=64, BK=64. Wave (wm 0..3, wn 0..1) owns 16 rows x 32 cols
// (two 16x16 accumulators) -> 4 wmma per K-block with A-fragment reuse.
// ---------------------------------------------------------------------------
#define TILE_M 64
#define TILE_N 64
#define TILE_K 64
#define LDSS   72   // row stride (elements); 144B = 9*16B keeps 16B alignment

__global__ __launch_bounds__(256)
void gemm_bf16_wmma_kernel(const float* __restrict__ A,
                           const float* __restrict__ B,
                           const float* __restrict__ bias,
                           float* __restrict__ C,
                           int M, int K, int Nn, unsigned flags)
{
    __shared__ __align__(16) __bf16 As[TILE_M][LDSS];
    __shared__ __align__(16) __bf16 Bs[TILE_N][LDSS];   // transposed: [n][k]

    const int tid  = threadIdx.x;
    const int wid  = tid >> 5;
    const int lane = tid & 31;
    const int wm   = wid & 3;       // wave row 0..3  (16 rows each)
    const int wn   = wid >> 2;      // wave col 0..1  (32 cols each)
    const int rowBase = blockIdx.y * TILE_M;
    const int colBase = blockIdx.x * TILE_N;
    const int lm  = lane & 15;
    const int lhi = lane >> 4;

    // staging coordinates (block-uniform mapping)
    const int ar_ = tid >> 2;            // A row 0..63
    const int ak0 = (tid & 3) * 16;      // A k-offset {0,16,32,48}
    const int bk0 = (tid >> 4) * 4;      // B k-offset 0..60
    const int bn0 = (tid & 15) * 4;      // B n-offset 0..60

    const bool fullMN = (rowBase + TILE_M <= M) && (colBase + TILE_N <= Nn);

    v8f acc0 = {0.f,0.f,0.f,0.f,0.f,0.f,0.f,0.f};
    v8f acc1 = {0.f,0.f,0.f,0.f,0.f,0.f,0.f,0.f};

    for (int kk = 0; kk < K; kk += TILE_K) {
        if (fullMN && (kk + TILE_K <= K)) {
            // ---- fast path: fully in-bounds, vectorized ----
            // A: 16 floats/thread as 4x float4, packed to 2x v8bf stores
            {
                const float* ap = A + (size_t)(rowBase + ar_) * K + kk + ak0;
                #pragma unroll
                for (int u = 0; u < 2; ++u) {
                    float4 f0 = *(const float4*)(ap + u * 8);
                    float4 f1 = *(const float4*)(ap + u * 8 + 4);
                    v8bf pk;
                    pk[0] = (__bf16)f0.x; pk[1] = (__bf16)f0.y;
                    pk[2] = (__bf16)f0.z; pk[3] = (__bf16)f0.w;
                    pk[4] = (__bf16)f1.x; pk[5] = (__bf16)f1.y;
                    pk[6] = (__bf16)f1.z; pk[7] = (__bf16)f1.w;
                    *(v8bf*)&As[ar_][ak0 + u * 8] = pk;
                }
            }
            // B: 4x4 block, register micro-transpose, v4bf stores into Bs[n][k]
            {
                float colv[4][4];
                #pragma unroll
                for (int i = 0; i < 4; ++i) {
                    float4 f = *(const float4*)(B + (size_t)(kk + bk0 + i) * Nn
                                                + colBase + bn0);
                    colv[0][i] = f.x; colv[1][i] = f.y;
                    colv[2][i] = f.z; colv[3][i] = f.w;
                }
                #pragma unroll
                for (int j = 0; j < 4; ++j) {
                    v4bf pk;
                    pk[0] = (__bf16)colv[j][0]; pk[1] = (__bf16)colv[j][1];
                    pk[2] = (__bf16)colv[j][2]; pk[3] = (__bf16)colv[j][3];
                    *(v4bf*)&Bs[bn0 + j][bk0] = pk;
                }
            }
        } else {
            // ---- slow path: clamp-and-multiply (branch-free loads) ----
            {
                int gr = rowBase + ar_;
                int cr = gr < M ? gr : M - 1;
                const float* ap = A + (size_t)cr * K;
                #pragma unroll
                for (int i = 0; i < 16; ++i) {
                    int gk = kk + ak0 + i;
                    int ck = gk < K ? gk : K - 1;
                    float msk = (gr < M && gk < K) ? 1.f : 0.f;
                    As[ar_][ak0 + i] = (__bf16)(ap[ck] * msk);
                }
            }
            {
                #pragma unroll
                for (int i = 0; i < 4; ++i) {
                    int gk = kk + bk0 + i;
                    int ck = gk < K ? gk : K - 1;
                    const float* bp = B + (size_t)ck * Nn;
                    #pragma unroll
                    for (int j = 0; j < 4; ++j) {
                        int gn = colBase + bn0 + j;
                        int cn = gn < Nn ? gn : Nn - 1;
                        float msk = (gk < K && gn < Nn) ? 1.f : 0.f;
                        Bs[bn0 + j][bk0 + i] = (__bf16)(bp[cn] * msk);
                    }
                }
            }
        }
        __syncthreads();

        #pragma unroll
        for (int s = 0; s < 2; ++s) {               // two K=32 steps
            const __bf16* arow = &As[wm * 16 + lm][s * 32];
            v8bf a0 = *(const v8bf*)(arow + lhi * 16);
            v8bf a1 = *(const v8bf*)(arow + lhi * 16 + 8);
            v16bf af = __builtin_shufflevector(a0, a1,
                0,1,2,3,4,5,6,7,8,9,10,11,12,13,14,15);

            const __bf16* brow0 = &Bs[wn * 32 + lm][s * 32];
            v8bf b0 = *(const v8bf*)(brow0 + lhi * 16);
            v8bf b1 = *(const v8bf*)(brow0 + lhi * 16 + 8);
            v16bf bf0 = __builtin_shufflevector(b0, b1,
                0,1,2,3,4,5,6,7,8,9,10,11,12,13,14,15);
            acc0 = __builtin_amdgcn_wmma_f32_16x16x32_bf16(
                false, af, false, bf0, (short)0, acc0, false, false);

            const __bf16* brow1 = &Bs[wn * 32 + 16 + lm][s * 32];
            v8bf b2 = *(const v8bf*)(brow1 + lhi * 16);
            v8bf b3 = *(const v8bf*)(brow1 + lhi * 16 + 8);
            v16bf bf1 = __builtin_shufflevector(b2, b3,
                0,1,2,3,4,5,6,7,8,9,10,11,12,13,14,15);
            acc1 = __builtin_amdgcn_wmma_f32_16x16x32_bf16(
                false, af, false, bf1, (short)0, acc1, false, false);
        }
        __syncthreads();
    }

    // epilogue: lane owns col n=lm of each 16-wide subtile,
    // rows wm*16 + lhi*8 + v (v=0..7)
    #pragma unroll
    for (int t2 = 0; t2 < 2; ++t2) {
        const v8f& acc = t2 ? acc1 : acc0;
        int gcol = colBase + wn * 32 + t2 * 16 + lm;
        if (gcol < Nn) {
            float bv = (flags & GF_BIAS) ? bias[gcol] : 0.f;
            #pragma unroll
            for (int v = 0; v < 8; ++v) {
                int grow = rowBase + wm * 16 + lhi * 8 + v;
                if (grow < M) {
                    size_t idx = (size_t)grow * Nn + gcol;
                    float r = acc[v] + bv;
                    if (flags & GF_ACC)  r += C[idx];
                    if (flags & GF_RELU) r = r > 0.f ? r : 0.f;
                    C[idx] = r;
                }
            }
        }
    }
}

// ---------------------------------------------------------------------------
// Embedding gather + concat:  x[n, 0:64]=emb[ids[n]], x[n, 64:96]=feats[n]
// ---------------------------------------------------------------------------
__global__ void embed_concat_kernel(const int* __restrict__ ids,
                                    const float* __restrict__ feats,
                                    const float* __restrict__ emb,
                                    float* __restrict__ x)
{
    int i = blockIdx.x * blockDim.x + threadIdx.x;
    if (i >= NN * (DD + F_IN)) return;
    int n = i / (DD + F_IN), j = i % (DD + F_IN);
    float v = (j < DD) ? emb[(size_t)ids[n] * DD + j]
                       : feats[(size_t)n * F_IN + (j - DD)];
    x[i] = v;
}

// ---------------------------------------------------------------------------
// GAT: el/er = sum_f h[n,h,f]*a{l,r}[h,f]   (warp per (node,head))
// ---------------------------------------------------------------------------
__global__ void elr_kernel(const float* __restrict__ h,
                           const float* __restrict__ al,
                           const float* __restrict__ ar,
                           float* __restrict__ el, float* __restrict__ er,
                           int Hh, int F)
{
    int node = blockIdx.x;
    int head = threadIdx.x >> 5;
    int lane = threadIdx.x & 31;
    const float* hp  = h  + (size_t)node * Hh * F + head * F;
    const float* alp = al + head * F;
    const float* arp = ar + head * F;
    float sl = 0.f, sr = 0.f;
    for (int i = lane; i < F; i += 32) {
        float v = hp[i];
        sl += v * alp[i];
        sr += v * arp[i];
    }
    #pragma unroll
    for (int off = 16; off > 0; off >>= 1) {
        sl += __shfl_down(sl, off, 32);
        sr += __shfl_down(sr, off, 32);
    }
    if (lane == 0) {
        el[node * Hh + head] = sl;
        er[node * Hh + head] = sr;
    }
}

// init out=bias broadcast, emax=-inf, den=0
__global__ void gat_init_kernel(float* __restrict__ out,
                                const float* __restrict__ bias,
                                float* __restrict__ emax,
                                float* __restrict__ den,
                                int HF, int Hh)
{
    size_t i = (size_t)blockIdx.x * blockDim.x + threadIdx.x;
    if (i < (size_t)NN * HF) out[i] = bias[i % HF];
    if (i < (size_t)NN * Hh) { emax[i] = -__builtin_inff(); den[i] = 0.f; }
}

__device__ __forceinline__ float leaky(float v) {
    return v > 0.f ? v : v * SLOPE;
}

// monotone float atomic max via int/uint punning
__device__ __forceinline__ void atomicMaxFloat(float* addr, float value) {
    if (value >= 0.f)
        atomicMax((int*)addr, __float_as_int(value));
    else
        atomicMin((unsigned int*)addr, __float_as_uint(value));
}

__global__ void edge_max_kernel(const int* __restrict__ src,
                                const int* __restrict__ dst,
                                const float* __restrict__ el,
                                const float* __restrict__ er,
                                float* __restrict__ emax,
                                int E, int Hh)
{
    int e = blockIdx.x * blockDim.x + threadIdx.x;
    if (e >= E) return;
    int s = src[e], d = dst[e];
    for (int hh = 0; hh < Hh; ++hh) {
        float v = leaky(el[s * Hh + hh] + er[d * Hh + hh]);
        atomicMaxFloat(&emax[d * Hh + hh], v);
    }
}

__global__ void edge_den_kernel(const int* __restrict__ src,
                                const int* __restrict__ dst,
                                const float* __restrict__ el,
                                const float* __restrict__ er,
                                const float* __restrict__ emax,
                                float* __restrict__ den,
                                int E, int Hh)
{
    int e = blockIdx.x * blockDim.x + threadIdx.x;
    if (e >= E) return;
    int s = src[e], d = dst[e];
    for (int hh = 0; hh < Hh; ++hh) {
        float v = leaky(el[s * Hh + hh] + er[d * Hh + hh]);
        atomicAdd(&den[d * Hh + hh], __expf(v - emax[d * Hh + hh]));
    }
}

// warp per edge: out[dst] += h[src] * alpha
__global__ void edge_agg_kernel(const int* __restrict__ src,
                                const int* __restrict__ dst,
                                const float* __restrict__ el,
                                const float* __restrict__ er,
                                const float* __restrict__ emax,
                                const float* __restrict__ den,
                                const float* __restrict__ h,
                                float* __restrict__ out,
                                int E, int Hh, int F)
{
    int warp = (blockIdx.x * blockDim.x + threadIdx.x) >> 5;
    int lane = threadIdx.x & 31;
    if (warp >= E) return;
    int s = src[warp], d = dst[warp];
    const float* hs = h   + (size_t)s * Hh * F;
    float*       od = out + (size_t)d * Hh * F;
    for (int hh = 0; hh < Hh; ++hh) {
        float v = leaky(el[s * Hh + hh] + er[d * Hh + hh]);
        float a = __expf(v - emax[d * Hh + hh]) / den[d * Hh + hh];
        for (int j = lane; j < F; j += 32)
            atomicAdd(&od[hh * F + j], hs[hh * F + j] * a);
    }
}

// ---------------------------------------------------------------------------
// Host-side helpers
// ---------------------------------------------------------------------------
static void gemm(const float* A, const float* B, const float* bias, float* C,
                 int M, int K, int Nn, unsigned flags, hipStream_t s)
{
    dim3 g((Nn + TILE_N - 1) / TILE_N, (M + TILE_M - 1) / TILE_M);
    gemm_bf16_wmma_kernel<<<g, 256, 0, s>>>(A, B, bias, C, M, K, Nn, flags);
}

static void gat(const float* x, const int* src, const int* dst,
                const float* W, const float* al, const float* ar,
                const float* b, float* hbuf,
                float* el, float* er, float* emax, float* den,
                float* out, int E, int K, int Hh, int F, hipStream_t s)
{
    int HF = Hh * F;
    gemm(x, W, nullptr, hbuf, NN, K, HF, 0, s);
    elr_kernel<<<NN, 32 * Hh, 0, s>>>(hbuf, al, ar, el, er, Hh, F);
    gat_init_kernel<<<((size_t)NN * HF + 255) / 256, 256, 0, s>>>(out, b, emax, den, HF, Hh);
    edge_max_kernel<<<(E + 255) / 256, 256, 0, s>>>(src, dst, el, er, emax, E, Hh);
    edge_den_kernel<<<(E + 255) / 256, 256, 0, s>>>(src, dst, el, er, emax, den, E, Hh);
    edge_agg_kernel<<<((size_t)E * 32 + 255) / 256, 256, 0, s>>>(
        src, dst, el, er, emax, den, hbuf, out, E, Hh, F);
}

extern "C" void kernel_launch(void* const* d_in, const int* in_sizes, int n_in,
                              void* d_out, int out_size, void* d_ws, size_t ws_size,
                              hipStream_t stream)
{
    // ---- inputs (setup_inputs dict order, params flattened) ----
    const float* feats_o = (const float*)d_in[0];
    const float* feats_t = (const float*)d_in[1];
    const int*   ids_o   = (const int*)d_in[2];
    const int*   ids_t   = (const int*)d_in[3];
    const int*   src_o   = (const int*)d_in[4];
    const int*   dst_o   = (const int*)d_in[5];
    const int*   src_t   = (const int*)d_in[6];
    const int*   dst_t   = (const int*)d_in[7];
    const float* emb     = (const float*)d_in[8];
    const float* mW1 = (const float*)d_in[9];  const float* mb1 = (const float*)d_in[10];
    const float* mW2 = (const float*)d_in[11]; const float* mb2 = (const float*)d_in[12];
    const float* mW3 = (const float*)d_in[13]; const float* mb3 = (const float*)d_in[14];
    const float* g1W = (const float*)d_in[15];
    const float* g1al = (const float*)d_in[16]; const float* g1ar = (const float*)d_in[17];
    const float* g1b  = (const float*)d_in[18];
    const float* g2W = (const float*)d_in[19];
    const float* g2al = (const float*)d_in[20]; const float* g2ar = (const float*)d_in[21];
    const float* g2b  = (const float*)d_in[22];
    const float* aW1 = (const float*)d_in[23]; const float* ab1 = (const float*)d_in[24];
    const float* aW2 = (const float*)d_in[25]; const float* ab2 = (const float*)d_in[26];
    const float* aW3 = (const float*)d_in[27]; const float* ab3 = (const float*)d_in[28];
    const float* tW1 = (const float*)d_in[29]; const float* tb1 = (const float*)d_in[30];
    const float* tW2 = (const float*)d_in[31]; const float* tb2 = (const float*)d_in[32];
    const float* tW3 = (const float*)d_in[33]; const float* tb3 = (const float*)d_in[34];

    const int E = in_sizes[4];   // edges (N*DEG + N)

    // ---- workspace bump allocation (floats) ----
    float* w = (float*)d_ws;
    float* E_O = w;  w += (size_t)NN * DD;         // 65536*64
    float* E_T = w;  w += (size_t)NN * DD;
    float* H1O = w;  w += (size_t)NN * HH_ * OUT1; // 65536*256
    float* H1T = w;  w += (size_t)NN * HH_ * OUT1;
    float* HB  = w;  w += (size_t)NN * HH_ * OUT1; // GAT hbuf scratch (max 256 wide)
    float* EL  = w;  w += (size_t)NN * HH_;
    float* ER  = w;  w += (size_t)NN * HH_;
    float* EM  = w;  w += (size_t)NN * HH_;
    float* DEN = w;  w += (size_t)NN * HH_;
    float* SCR = w;  w += (size_t)NN * 352;        // transient region
    // aliases inside SCR:
    float* XBUF = SCR;                             // 65536*96   (embed input)
    float* M1   = SCR + (size_t)NN * 96;           // 65536*128
    float* M2   = SCR + (size_t)NN * 224;          // 65536*128
    float* H2O  = SCR;                             // 65536*128  (after MLP dead)
    float* H2T  = SCR + (size_t)NN * 128;          // 65536*128
    float* TMP1 = SCR + (size_t)NN * 256;          // >= 128*500
    float* TMP2 = TMP1 + 128 * 512;                // >= 128*100

    float* traff_out  = (float*)d_out;             // (128, 512)
    float* routes_out = (float*)d_out + BB * NEDGES; // (128, 64)

    const int XDIM = DD + F_IN;  // 96
    dim3 cg(((size_t)NN * XDIM + 255) / 256);

    // ---- EdgeEmbedding (online) ----
    embed_concat_kernel<<<cg, 256, 0, stream>>>(ids_o, feats_o, emb, XBUF);
    gemm(XBUF, mW1, mb1, M1, NN, XDIM, HID, GF_BIAS | GF_RELU, stream);
    gemm(M1,   mW2, mb2, M2, NN, HID,  HID, GF_BIAS | GF_RELU, stream);
    gemm(M2,   mW3, mb3, E_O, NN, HID, DD,  GF_BIAS, stream);
    // ---- EdgeEmbedding (target) ----
    embed_concat_kernel<<<cg, 256, 0, stream>>>(ids_t, feats_t, emb, XBUF);
    gemm(XBUF, mW1, mb1, M1, NN, XDIM, HID, GF_BIAS | GF_RELU, stream);
    gemm(M1,   mW2, mb2, M2, NN, HID,  HID, GF_BIAS | GF_RELU, stream);
    gemm(M2,   mW3, mb3, E_T, NN, HID, DD,  GF_BIAS, stream);

    // ---- GATConv 1 (o on graph o, t on graph t) ----
    gat(E_O, src_o, dst_o, g1W, g1al, g1ar, g1b, HB, EL, ER, EM, DEN,
        H1O, E, DD, HH_, OUT1, stream);
    gat(E_T, src_t, dst_t, g1W, g1al, g1ar, g1b, HB, EL, ER, EM, DEN,
        H1T, E, DD, HH_, OUT1, stream);

    // ---- GATConv 2 (both on ONLINE graph, per reference) ----
    gat(H1O, src_o, dst_o, g2W, g2al, g2ar, g2b, HB, EL, ER, EM, DEN,
        H2O, E, HH_ * OUT1, HH_, DD, stream);
    gat(H1T, src_o, dst_o, g2W, g2al, g2ar, g2b, HB, EL, ER, EM, DEN,
        H2T, E, HH_ * OUT1, HH_, DD, stream);

    // H2O/H2T are (N, H*D) row-major == (B, in_size) row-major.

    // ---- traffic head: h2o -> 300 -> 100 -> 512 ----
    gemm(H2O,  tW1, tb1, TMP1, BB, IN_SIZE, 300, GF_BIAS | GF_RELU, stream);
    gemm(TMP1, tW2, tb2, TMP2, BB, 300,     100, GF_BIAS | GF_RELU, stream);
    gemm(TMP2, tW3, tb3, traff_out, BB, 100, NEDGES, GF_BIAS, stream);

    // ---- action head: [h2o|h2t] -> 500 -> 100 -> 64 (split-K over concat) ----
    gemm(H2O,  aW1,                          nullptr, TMP1, BB, IN_SIZE, 500, 0, stream);
    gemm(H2T,  aW1 + (size_t)IN_SIZE * 500,  ab1,     TMP1, BB, IN_SIZE, 500,
         GF_ACC | GF_BIAS | GF_RELU, stream);
    gemm(TMP1, aW2, ab2, TMP2, BB, 500, 100, GF_BIAS | GF_RELU, stream);
    gemm(TMP2, aW3, ab3, routes_out, BB, 100, 64, GF_BIAS, stream);
}
